// NeuronCircuitDown_31593779429534
// MI455X (gfx1250) — matrix operand order repaired and checked by reading, not compiled, not run on hardware
//
#include <hip/hip_runtime.h>
#include <hip/hip_bf16.h>

typedef __attribute__((ext_vector_type(16))) __bf16 v16bf;
typedef __attribute__((ext_vector_type(8)))  __bf16 v8bf;
typedef __attribute__((ext_vector_type(8)))  float  v8f;

#define BB 4
#define SS 4096
#define DD 1024
#define RR 128
#define NIN 8
#define NPROC 32
#define KK 4
#define MTOT (BB*SS) /* 16384 tokens */

// ---------------------------------------------------------------------------
// Prep 1: Wt[n][r][d] = bf16(input_neurons[n][d][r])  (K=D contiguous so the
// B-matrix fragment is a single 32B contiguous load per lane)
// ---------------------------------------------------------------------------
__global__ __launch_bounds__(256) void prep_wt_kernel(const float* __restrict__ w_in,
                                                      __bf16* __restrict__ wt) {
    int i = blockIdx.x * blockDim.x + threadIdx.x;   // 0 .. NIN*RR*DD-1
    int d = i & (DD - 1);
    int t = i >> 10;          // n*RR + r
    int r = t & (RR - 1);
    int n = t >> 7;
    wt[i] = (__bf16)w_in[(n * DD + d) * RR + r];
}

// ---------------------------------------------------------------------------
// Prep 2: xbf = bf16(x), row-major [MTOT][DD].  8 elements per thread.
// ---------------------------------------------------------------------------
__global__ __launch_bounds__(256) void prep_xbf_kernel(const float* __restrict__ x,
                                                       __bf16* __restrict__ xbf) {
    const size_t i = (size_t)(blockIdx.x * blockDim.x + threadIdx.x) * 8;
    float4 a = *(const float4*)(x + i);
    float4 b = *(const float4*)(x + i + 4);
    v8bf o;
    o[0] = (__bf16)a.x; o[1] = (__bf16)a.y; o[2] = (__bf16)a.z; o[3] = (__bf16)a.w;
    o[4] = (__bf16)b.x; o[5] = (__bf16)b.y; o[6] = (__bf16)b.z; o[7] = (__bf16)b.w;
    *(v8bf*)(xbf + i) = o;
}

// ---------------------------------------------------------------------------
// Projection: Y_n = xbf @ Wt[n]  (8 accumulators per M-tile, unscaled A shared
// across n), then epilogue y[m,r] = sum_n iw[m,n] * Y_n[m,r].
// Wave = one 16-wide r tile x two 16-row M tiles. Block = 8 waves = full R.
// ---------------------------------------------------------------------------
__global__ __launch_bounds__(256) void proj_kernel(const __bf16* __restrict__ xbf,
                                                   const float* __restrict__ iw,
                                                   const __bf16* __restrict__ wt,
                                                   float* __restrict__ y) {
    const int lane      = threadIdx.x & 31;
    const int rt        = threadIdx.x >> 5;   // 0..7 : which 16-wide r tile
    const int rowInTile = lane & 15;
    const int hi        = lane >> 4;          // upper half-wave flag
    const int mbase     = blockIdx.x * 32;

    v8f acc[2][NIN] = {};                     // [m-tile][n] 16x16 f32 C tiles

    const int koffA = hi * 16;                // A: lanes16-31 hold K=16..31
    const int koffB = hi * 16;                // B: lanes16-31 hold K=16..31
    const int colB  = rt * 16 + rowInTile;    // B column (N index)

    const __bf16* pA0 = xbf + (size_t)(mbase + rowInTile) * DD + koffA;
    const __bf16* pA1 = pA0 + (size_t)16 * DD;
    const __bf16* pB  = wt + (size_t)colB * DD + koffB;

    for (int d0 = 0; d0 < DD; d0 += 32) {
        // A fragments: K halves {0..7} and {8..15} (+koffA), merged reg-to-reg
        v8bf a0l = *(const v8bf*)(pA0 + d0);
        v8bf a0h = *(const v8bf*)(pA0 + d0 + 8);
        v8bf a1l = *(const v8bf*)(pA1 + d0);
        v8bf a1h = *(const v8bf*)(pA1 + d0 + 8);
        v16bf a0 = __builtin_shufflevector(a0l, a0h, 0,1,2,3,4,5,6,7,8,9,10,11,12,13,14,15);
        v16bf a1 = __builtin_shufflevector(a1l, a1h, 0,1,2,3,4,5,6,7,8,9,10,11,12,13,14,15);
        #pragma unroll
        for (int n = 0; n < NIN; ++n) {
            v16bf bfrag = *(const v16bf*)(pB + (size_t)n * RR * DD + d0);
            acc[0][n] = __builtin_amdgcn_wmma_f32_16x16x32_bf16(
                false, a0, false, bfrag, (short)0, acc[0][n], false, false);
            acc[1][n] = __builtin_amdgcn_wmma_f32_16x16x32_bf16(
                false, a1, false, bfrag, (short)0, acc[1][n], false, false);
        }
    }

    // Epilogue: per-row weighted sum over n.
    // C layout: element j -> row (j + hi*8), col = rt*16 + (lane&15)
    #pragma unroll
    for (int mt = 0; mt < 2; ++mt) {
        #pragma unroll
        for (int j = 0; j < 8; ++j) {
            const int row = mbase + mt * 16 + j + hi * 8;
            const float4* pw = (const float4*)(iw + (size_t)row * NIN);
            const float4 wa = pw[0], wb = pw[1];
            float s = wa.x * acc[mt][0][j] + wa.y * acc[mt][1][j]
                    + wa.z * acc[mt][2][j] + wa.w * acc[mt][3][j]
                    + wb.x * acc[mt][4][j] + wb.y * acc[mt][5][j]
                    + wb.z * acc[mt][6][j] + wb.w * acc[mt][7][j];
            y[(size_t)row * RR + rt * 16 + rowInTile] = s;
        }
    }
}

// ---------------------------------------------------------------------------
// Householder chain: one wave per token; lane holds float4 (R=128 = 32*4).
// x <- x - 2*(v.x)/(|v|^2+1e-8) * v   (== reference's normalized form)
// ---------------------------------------------------------------------------
__global__ __launch_bounds__(256) void householder_kernel(const float* __restrict__ y,
                                                          const int* __restrict__ pidx,
                                                          const float* __restrict__ pn,
                                                          float* __restrict__ out) {
    const int lane  = threadIdx.x & 31;
    const int wave  = threadIdx.x >> 5;
    const int token = blockIdx.x * 8 + wave;

    float4 xc = ((const float4*)(y + (size_t)token * RR))[lane];

    #pragma unroll
    for (int k = 0; k < KK; ++k) {
        const int idx = pidx[token * KK + k];
        float4 v = ((const float4*)(pn + (size_t)idx * RR))[lane];
        float nsq = v.x * v.x + v.y * v.y + v.z * v.z + v.w * v.w;
        float dot = v.x * xc.x + v.y * xc.y + v.z * xc.z + v.w * xc.w;
        #pragma unroll
        for (int off = 16; off >= 1; off >>= 1) {
            nsq += __shfl_xor(nsq, off, 32);
            dot += __shfl_xor(dot, off, 32);
        }
        const float s = 2.0f * dot / (nsq + 1e-8f);
        xc.x -= s * v.x; xc.y -= s * v.y; xc.z -= s * v.z; xc.w -= s * v.w;
    }
    ((float4*)(out + (size_t)token * RR))[lane] = xc;
}

// ---------------------------------------------------------------------------
extern "C" void kernel_launch(void* const* d_in, const int* in_sizes, int n_in,
                              void* d_out, int out_size, void* d_ws, size_t ws_size,
                              hipStream_t stream) {
    const float* x    = (const float*)d_in[0];   // [B,S,D]
    const float* iw   = (const float*)d_in[1];   // [B,S,NIN]
    const int*   pidx = (const int*)  d_in[2];   // [B,S,KK]
    const float* inn  = (const float*)d_in[3];   // [NIN,D,R]
    const float* pn   = (const float*)d_in[4];   // [NPROC,R]

    char* ws = (char*)d_ws;
    __bf16* wt  = (__bf16*)ws;                                   // 2 MB
    __bf16* xbf = (__bf16*)(ws + (size_t)NIN * RR * DD * 2);     // 33.5 MB
    float*  yb  = (float*)(ws + (size_t)NIN * RR * DD * 2
                              + (size_t)MTOT * DD * 2);          // 8 MB

    prep_wt_kernel <<<(NIN * RR * DD) / 256, 256, 0, stream>>>(inn, wt);
    prep_xbf_kernel<<<((size_t)MTOT * DD) / (8 * 256), 256, 0, stream>>>(x, xbf);
    proj_kernel    <<<MTOT / 32, 256, 0, stream>>>(xbf, iw, wt, yb);
    householder_kernel<<<MTOT / 8, 256, 0, stream>>>(yb, pidx, pn, (float*)d_out);
}
